// PointCloudTCN_44392781971605
// MI455X (gfx1250) — matrix operand order repaired and checked by reading, not compiled, not run on hardware
//
#include <hip/hip_runtime.h>
#include <hip/hip_bf16.h>

// ---------------------------------------------------------------------------
// PointCloudTCN on MI455X (gfx1250, wave32, WMMA).
//
// Roofline: ~170 GFLOP of small-K GEMMs over 262144 edge rows vs ~0.5 GB of
// traffic @ 23.3 TB/s. All Linear layers run on v_wmma_f32_16x16x32_f16
// (f16 in, f32 accumulate). W^T (max 384x128 = 96KB f16) is staged in LDS
// (320KB/WGP) via GLOBAL_LOAD_ASYNC_TO_LDS_B128 (ASYNCcnt path) and shared by
// 8 waves; each wave owns a 32-row M-tile so every B fragment fetched from
// LDS feeds two WMMAs (2x arithmetic intensity on the ds path, two
// independent accumulators to cover WMMA->WMMA RAW hazards).
// ---------------------------------------------------------------------------

typedef __attribute__((ext_vector_type(16))) _Float16 v16h;
typedef __attribute__((ext_vector_type(8)))  _Float16 v8h;
typedef __attribute__((ext_vector_type(8)))  float    v8f;
typedef __attribute__((ext_vector_type(4)))  int      v4i;

#define N_PTS  16384
#define NI     14
#define H_DIM  64
#define E_DIM  64
#define HIDD   128
#define HO     8
#define KNN    16
#define LHC    3
#define NEDGE  (N_PTS * KNN)   // 262144

// Async global->LDS staging (gfx1250 ASYNCcnt path) if the toolchain has it.
#if defined(__has_builtin)
#if __has_builtin(__builtin_amdgcn_global_load_async_to_lds_b128) && \
    __has_builtin(__builtin_amdgcn_s_wait_asynccnt)
#define USE_ASYNC_LDS 1
#endif
#endif

#ifdef USE_ASYNC_LDS
typedef __attribute__((address_space(1))) v4i gbl_v4i;   // global (64-bit ptr)
typedef __attribute__((address_space(3))) v4i lds_v4i;   // LDS (32-bit ptr)
__device__ __forceinline__ gbl_v4i* as_gbl(const void* p) {
  return (gbl_v4i*)(unsigned long long)p;
}
__device__ __forceinline__ lds_v4i* as_lds(void* p) {
  // low 32 bits of a flat shared pointer = LDS byte offset (aperture rule)
  return (lds_v4i*)(unsigned int)(unsigned long long)p;
}
#endif

// ---------------------------------------------------------------------------
// WMMA fragment loaders (gfx1250 wave32 layouts, CDNA5 ISA 7.12.2)
// ---------------------------------------------------------------------------
// A 16x32 f16: lane L row M=L&15. halves 0..7 -> K = k0 + (L&16? 8:0) + 0..7
//                               halves 8..15 -> same + 16
__device__ __forceinline__ v16h load_a_frag(const _Float16* __restrict__ Xrows,
                                            int Kp, int lane, int k0) {
  const _Float16* p = Xrows + (size_t)(lane & 15) * Kp + k0 + ((lane & 16) ? 8 : 0);
  v8h lo = *(const v8h*)(p);
  v8h hi = *(const v8h*)(p + 16);
  v16h a;
#pragma unroll
  for (int i = 0; i < 8; ++i) { a[i] = lo[i]; a[8 + i] = hi[i]; }
  return a;
}

// B 32x16 f16 from LDS, W^T stored [Np x Kp] row-major:
// lane L -> column n = nt*16 + (L&15); halves 0..7 -> K = k0 + (L&16?8:0) + h, halves 8..15 -> same + 16
__device__ __forceinline__ v16h load_b_frag(const _Float16* __restrict__ WtLds,
                                            int Kp, int lane, int nt, int k0) {
  const _Float16* p = WtLds + (size_t)((nt << 4) + (lane & 15)) * Kp
                            + k0 + ((lane & 16) ? 8 : 0);
  v8h lo = *(const v8h*)(p);
  v8h hi = *(const v8h*)(p + 16);
  v16h b;
#pragma unroll
  for (int i = 0; i < 8; ++i) { b[i] = lo[i]; b[8 + i] = hi[i]; }
  return b;
}

// ---------------------------------------------------------------------------
// Fused Linear(+bias)(+ReLU):  Y[M x Np] = act(X[M x KP] * W + b)
// 256 threads = 8 waves; wave owns 32 rows (two 16-row tiles sharing each B
// fragment). M multiple of 256, KP multiple of 32, Np multiple of 16.
// ---------------------------------------------------------------------------
template <int KP>
__global__ void k_gemm(const _Float16* __restrict__ X,
                       const _Float16* __restrict__ Wt,   // [Np x KP] f16
                       const float* __restrict__ bias,    // [Nreal] f32
                       _Float16* __restrict__ Yh,         // f16 out (or null)
                       float* __restrict__ Yf,            // f32 out (or null)
                       int M, int Np, int Nreal, int relu) {
  extern __shared__ __align__(64) _Float16 lds_w[];
  constexpr int KSTEPS = KP / 32;
  const int tid  = threadIdx.x;
  const int lane = tid & 31;
  const int wave = tid >> 5;

  const int rowbase = blockIdx.x * 256 + wave * 32;
  const _Float16* Xbase = X + (size_t)rowbase * KP;

  // ---- A fragments for both 16-row tiles: register-resident for whole tile
  v16h a0[KSTEPS], a1[KSTEPS];
#pragma unroll
  for (int ks = 0; ks < KSTEPS; ++ks) {
    a0[ks] = load_a_frag(Xbase, KP, lane, ks * 32);
    a1[ks] = load_a_frag(Xbase + (size_t)16 * KP, KP, lane, ks * 32);
  }

  // ---- stage W^T into LDS ----
  const int tot16 = (KP * Np) >> 3;  // in 16B units
#ifdef USE_ASYNC_LDS
  {
    const v8h* wsrc = (const v8h*)Wt;
    v8h* wdst = (v8h*)lds_w;
    for (int i = tid; i < tot16; i += blockDim.x)
      __builtin_amdgcn_global_load_async_to_lds_b128(as_gbl(wsrc + i),
                                                     as_lds(wdst + i), 0, 0);
    __builtin_amdgcn_s_wait_asynccnt(0);
  }
#else
  {
    const v8h* wsrc = (const v8h*)Wt;
    v8h* wdst = (v8h*)lds_w;
    for (int i = tid; i < tot16; i += blockDim.x) wdst[i] = wsrc[i];
  }
#endif
  __syncthreads();
  if (rowbase >= M) return;

  for (int nt = 0; nt < (Np >> 4); ++nt) {
    v8f acc0 = {}, acc1 = {};
#pragma unroll
    for (int ks = 0; ks < KSTEPS; ++ks) {
      v16h b = load_b_frag(lds_w, KP, lane, nt, ks * 32);
      acc0 = __builtin_amdgcn_wmma_f32_16x16x32_f16(
          false, a0[ks], false, b, (short)0, acc0, false, false);
      acc1 = __builtin_amdgcn_wmma_f32_16x16x32_f16(
          false, a1[ks], false, b, (short)0, acc1, false, false);
    }
    // C/D layout: lane L col n = L&15, VGPR r -> row r + (L&16 ? 8 : 0)
    const int n  = (nt << 4) + (lane & 15);
    const float bv = (n < Nreal) ? bias[n] : 0.0f;
    const int mo = (lane & 16) ? 8 : 0;
#pragma unroll
    for (int r = 0; r < 8; ++r) {
      float v0 = acc0[r] + bv;
      float v1 = acc1[r] + bv;
      if (relu) { v0 = v0 > 0.0f ? v0 : 0.0f; v1 = v1 > 0.0f ? v1 : 0.0f; }
      const size_t off0 = (size_t)(rowbase + mo + r) * Np + n;
      const size_t off1 = (size_t)(rowbase + 16 + mo + r) * Np + n;
      if (Yh) { Yh[off0] = (_Float16)v0; Yh[off1] = (_Float16)v1; }
      else    { Yf[off0] = v0;           Yf[off1] = v1; }
    }
  }
}

// ---------------------------------------------------------------------------
// Weight prep: W[K x N] f32 -> W^T padded [Np x Kp] f16 (zero pad)
// ---------------------------------------------------------------------------
__global__ void k_wprep(const float* __restrict__ W, _Float16* __restrict__ Wt,
                        int K, int Nn, int Kp, int Np) {
  const int idx = blockIdx.x * 256 + threadIdx.x;
  if (idx >= Kp * Np) return;
  const int n = idx / Kp, k = idx % Kp;
  const float v = (n < Nn && k < K) ? W[(size_t)k * Nn + n] : 0.0f;
  Wt[(size_t)n * Kp + k] = (_Float16)v;
}

// ---------------------------------------------------------------------------
// Brute-force KNN (k=16): one thread per query, LDS-tiled candidates,
// register-resident insertion-sorted top-16.
// ---------------------------------------------------------------------------
__global__ void k_knn(const float* __restrict__ x, int* __restrict__ srcIdx) {
  __shared__ float tile[128 * 16];
  const int i = blockIdx.x * 128 + threadIdx.x;
  float q[NI];
#pragma unroll
  for (int f = 0; f < NI; ++f) q[f] = x[(size_t)i * NI + f];
  float bd[KNN]; int bi[KNN];
#pragma unroll
  for (int j = 0; j < KNN; ++j) { bd[j] = 3.4e38f; bi[j] = 0; }

  for (int t0 = 0; t0 < N_PTS; t0 += 128) {
    __syncthreads();
    for (int u = threadIdx.x; u < 128 * NI; u += 128) {
      const int c = u / NI, f = u % NI;
      tile[c * 16 + f] = x[(size_t)(t0 + c) * NI + f];
    }
    __syncthreads();
    for (int c = 0; c < 128; ++c) {
      const int j = t0 + c;
      float d = 0.0f;
#pragma unroll
      for (int f = 0; f < NI; ++f) {
        const float df = q[f] - tile[c * 16 + f];
        d = fmaf(df, df, d);
      }
      if (j == i) d = 3.4e38f;   // exclude self-loop
      if (d < bd[KNN - 1]) {
        bd[KNN - 1] = d; bi[KNN - 1] = j;
#pragma unroll
        for (int s = KNN - 1; s > 0; --s) {
          if (bd[s] < bd[s - 1]) {
            float td = bd[s]; bd[s] = bd[s - 1]; bd[s - 1] = td;
            int   ti = bi[s]; bi[s] = bi[s - 1]; bi[s - 1] = ti;
          }
        }
      }
    }
  }
#pragma unroll
  for (int j = 0; j < KNN; ++j) srcIdx[i * KNN + j] = bi[j];
}

// ---------------------------------------------------------------------------
// Elementwise / gather / segment kernels. Edges of node i are e = i*16..i*16+15.
// ---------------------------------------------------------------------------
__global__ void k_build_enc_in(const float* __restrict__ x,
                               const int* __restrict__ src,
                               _Float16* __restrict__ out) {  // [E x 32]
  const int idx = blockIdx.x * 256 + threadIdx.x;
  const int e = idx >> 5, c = idx & 31;
  const int i = e >> 4;
  const int j = src[e];
  float v = 0.0f;
  if (c < NI) v = x[(size_t)i * NI + c];
  else if (c < 2 * NI) { const int f = c - NI; v = x[(size_t)j * NI + f] - x[(size_t)i * NI + f]; }
  out[idx] = (_Float16)v;
}

__global__ void k_segmax_relu(const _Float16* __restrict__ msg,  // [E x 64]
                              float* __restrict__ hf, _Float16* __restrict__ hh) {
  const int idx = blockIdx.x * 256 + threadIdx.x;  // N*64
  const int i = idx >> 6, f = idx & 63;
  float m = -3.4e38f;
#pragma unroll
  for (int t = 0; t < KNN; ++t)
    m = fmaxf(m, (float)msg[(size_t)(i * KNN + t) * 64 + f]);
  m = fmaxf(m, 0.0f);  // relu(segment_max)
  hf[idx] = m; hh[idx] = (_Float16)m;
}

__global__ void k_build_edge_in(const _Float16* __restrict__ h,
                                const int* __restrict__ src,
                                _Float16* __restrict__ out) {  // [E x 128] = [h_src|h_dst]
  const int idx = blockIdx.x * 256 + threadIdx.x;
  const int e = idx >> 7, c = idx & 127;
  const int i = e >> 4;
  const int j = src[e];
  out[idx] = (c < 64) ? h[(size_t)j * 64 + c] : h[(size_t)i * 64 + (c - 64)];
}

__global__ void k_build_rel_in(const _Float16* __restrict__ h,
                               const _Float16* __restrict__ eprev,
                               const int* __restrict__ src,
                               _Float16* __restrict__ out) {  // [E x 192] = [h_dst|h_src|e]
  const int idx = blockIdx.x * 256 + threadIdx.x;
  const int e = idx / 192, c = idx % 192;
  const int i = e >> 4;
  _Float16 v;
  if (c < 64)        v = h[(size_t)i * 64 + c];
  else if (c < 128)  v = h[(size_t)src[e] * 64 + (c - 64)];
  else               v = eprev[(size_t)e * 64 + (c - 128)];
  out[idx] = v;
}

__global__ void k_build_obj_in(const _Float16* __restrict__ h,
                               const _Float16* __restrict__ ecur,
                               _Float16* __restrict__ out) {  // [N x 128] = [h|segsum(e)]
  const int idx = blockIdx.x * 256 + threadIdx.x;
  const int i = idx >> 7, c = idx & 127;
  if (c < 64) { out[idx] = h[(size_t)i * 64 + c]; return; }
  const int f = c - 64;
  float s = 0.0f;
#pragma unroll
  for (int t = 0; t < KNN; ++t)
    s += (float)ecur[(size_t)(i * KNN + t) * 64 + f];
  out[idx] = (_Float16)s;
}

__global__ void k_blend(float* __restrict__ hf, const float* __restrict__ dh,
                        _Float16* __restrict__ hh) {  // h = 0.5h + 0.5*delta
  const int idx = blockIdx.x * 256 + threadIdx.x;  // N*64
  const float v = 0.5f * hf[idx] + 0.5f * dh[idx];
  hf[idx] = v; hh[idx] = (_Float16)v;
}

__global__ void k_build_prel_in(const _Float16* __restrict__ h,
                                const _Float16* __restrict__ e0,
                                const _Float16* __restrict__ e1,
                                const _Float16* __restrict__ e2,
                                const _Float16* __restrict__ e3,
                                const int* __restrict__ src,
                                _Float16* __restrict__ out) { // [E x 384]
  const int idx = blockIdx.x * 256 + threadIdx.x;
  const int e = idx / 384, c = idx % 384;
  const int i = e >> 4;
  _Float16 v;
  if (c < 64)        v = h[(size_t)i * 64 + c];
  else if (c < 128)  v = h[(size_t)src[e] * 64 + (c - 64)];
  else {
    const int s = (c - 128) >> 6, f = (c - 128) & 63;
    const _Float16* eb = (s == 0) ? e0 : (s == 1) ? e1 : (s == 2) ? e2 : e3;
    v = eb[(size_t)e * 64 + f];
  }
  out[idx] = v;
}

__global__ void k_build_pobj_in(const _Float16* __restrict__ h,
                                const float* __restrict__ ep,  // [E x 16] f32, col 0
                                _Float16* __restrict__ out) {  // [N x 96]
  const int idx = blockIdx.x * 256 + threadIdx.x;
  const int i = idx / 96, c = idx % 96;
  float v = 0.0f;
  if (c < 64) v = (float)h[(size_t)i * 64 + c];
  else if (c == 64) {
    float s = 0.0f;
#pragma unroll
    for (int t = 0; t < KNN; ++t) s += ep[(size_t)(i * KNN + t) * 16];
    v = s;
  }
  out[idx] = (_Float16)v;
}

__global__ void k_sigmoid_beta(const float* __restrict__ bB, float* __restrict__ out) {
  const int i = blockIdx.x * 256 + threadIdx.x;  // N
  out[i] = 1.0f / (1.0f + __expf(-bB[(size_t)i * 16])) + 1e-5f;
}

__global__ void k_copy_hout(const float* __restrict__ xo, float* __restrict__ out) {
  const int idx = blockIdx.x * 256 + threadIdx.x;  // N*8
  const int i = idx >> 3, f = idx & 7;
  out[idx] = xo[(size_t)i * 16 + f];
}

__global__ void k_copy_col0(const float* __restrict__ t, float* __restrict__ out) {
  const int i = blockIdx.x * 256 + threadIdx.x;  // N
  out[i] = t[(size_t)i * 16];
}

// ---------------------------------------------------------------------------
// Host orchestration
// ---------------------------------------------------------------------------
static void launch_gemm_kp(int Kp, const _Float16* X, const _Float16* Wt,
                           const float* bias, _Float16* Yh, float* Yf,
                           int M, int Np, int Nreal, int relu, hipStream_t stream) {
  const size_t lds = (size_t)Kp * Np * 2;
  const dim3 grid(M / 256), block(256);
  switch (Kp) {
    case 32:  k_gemm<32> <<<grid, block, lds, stream>>>(X, Wt, bias, Yh, Yf, M, Np, Nreal, relu); break;
    case 64:  k_gemm<64> <<<grid, block, lds, stream>>>(X, Wt, bias, Yh, Yf, M, Np, Nreal, relu); break;
    case 96:  k_gemm<96> <<<grid, block, lds, stream>>>(X, Wt, bias, Yh, Yf, M, Np, Nreal, relu); break;
    case 128: k_gemm<128><<<grid, block, lds, stream>>>(X, Wt, bias, Yh, Yf, M, Np, Nreal, relu); break;
    case 192: k_gemm<192><<<grid, block, lds, stream>>>(X, Wt, bias, Yh, Yf, M, Np, Nreal, relu); break;
    default:  k_gemm<384><<<grid, block, lds, stream>>>(X, Wt, bias, Yh, Yf, M, Np, Nreal, relu); break;
  }
}

extern "C" void kernel_launch(void* const* d_in, const int* in_sizes, int n_in,
                              void* d_out, int out_size, void* d_ws, size_t ws_size,
                              hipStream_t stream) {
  (void)in_sizes; (void)n_in; (void)out_size; (void)ws_size;
  const float* x = (const float*)d_in[0];
  float* out = (float*)d_out;

  // ---- workspace bump allocator (256B aligned) ----
  char* ws = (char*)d_ws;
  size_t cur = 0;
  auto walloc = [&](size_t bytes) -> char* {
    cur = (cur + 255) & ~(size_t)255;
    char* p = ws + cur; cur += bytes; return p;
  };

  int*       srcIdx = (int*)walloc((size_t)NEDGE * 4);
  _Float16*  bufIn  = (_Float16*)walloc((size_t)NEDGE * 384 * 2);
  _Float16*  bufH1  = (_Float16*)walloc((size_t)NEDGE * 128 * 2);
  _Float16*  bufH2  = (_Float16*)walloc((size_t)NEDGE * 128 * 2);
  _Float16*  eBuf[LHC + 1];
  for (int i = 0; i <= LHC; ++i) eBuf[i] = (_Float16*)walloc((size_t)NEDGE * 64 * 2);
  float*     epF    = (float*)walloc((size_t)NEDGE * 16 * 4);
  float*     hF     = (float*)walloc((size_t)N_PTS * 64 * 4);
  _Float16*  hH     = (_Float16*)walloc((size_t)N_PTS * 64 * 2);
  _Float16*  nodeIn = (_Float16*)walloc((size_t)N_PTS * 128 * 2);
  _Float16*  nH1    = (_Float16*)walloc((size_t)N_PTS * 128 * 2);
  _Float16*  nH2    = (_Float16*)walloc((size_t)N_PTS * 128 * 2);
  float*     deltaF = (float*)walloc((size_t)N_PTS * 64 * 4);
  float*     bBF    = (float*)walloc((size_t)N_PTS * 16 * 4);
  float*     xoF    = (float*)walloc((size_t)N_PTS * 16 * 4);
  float*     trF    = (float*)walloc((size_t)N_PTS * 16 * 4);

  // ---- weight prep (f32 [KxN] -> padded f16 [Np x Kp] transposed) ----
  _Float16*    Wt[64]; const float* Bias[64];
  int KpA[64], NpA[64], NrA[64]; int nl = 0;
  auto prep = [&](int wi, int K, int Nn, int Kp, int Np) -> int {
    _Float16* wt = (_Float16*)walloc((size_t)Kp * Np * 2);
    const int total = Kp * Np;
    k_wprep<<<dim3((total + 255) / 256), dim3(256), 0, stream>>>(
        (const float*)d_in[wi], wt, K, Nn, Kp, Np);
    Wt[nl] = wt; Bias[nl] = (const float*)d_in[wi + 1];
    KpA[nl] = Kp; NpA[nl] = Np; NrA[nl] = Nn;
    return nl++;
  };
  auto gemm = [&](int lid, const _Float16* Xin, _Float16* Yh, float* Yf,
                  int M, int relu) {
    launch_gemm_kp(KpA[lid], Xin, Wt[lid], Bias[lid], Yh, Yf, M,
                   NpA[lid], NrA[lid], relu, stream);
  };

  // d_in layout (insertion-order, depth-first flattening of setup_inputs()):
  // 0:x | node_encoder 1..4 | edge_encoder 5..8 | hc[i] at 9+16i (rel 8, obj 8)
  // | B 57..64 | X 65..72 | P_rel 73..80 | P_obj 81..88
  const int L_enc0 = prep(1, 2 * NI, HIDD, 32, 128);
  const int L_enc1 = prep(3, HIDD, H_DIM, 128, 64);
  const int L_ee0  = prep(5, 2 * H_DIM, HIDD, 128, 128);
  const int L_ee1  = prep(7, HIDD, E_DIM, 128, 64);
  int L_rel[LHC][4], L_obj[LHC][4];
  for (int i = 0; i < LHC; ++i) {
    const int base = 9 + i * 16;
    L_rel[i][0] = prep(base + 0, 2 * H_DIM + E_DIM, HIDD, 192, 128);
    L_rel[i][1] = prep(base + 2, HIDD, HIDD, 128, 128);
    L_rel[i][2] = prep(base + 4, HIDD, HIDD, 128, 128);
    L_rel[i][3] = prep(base + 6, HIDD, E_DIM, 128, 64);
    L_obj[i][0] = prep(base + 8,  H_DIM + E_DIM, HIDD, 128, 128);
    L_obj[i][1] = prep(base + 10, HIDD, HIDD, 128, 128);
    L_obj[i][2] = prep(base + 12, HIDD, HIDD, 128, 128);
    L_obj[i][3] = prep(base + 14, HIDD, H_DIM, 128, 64);
  }
  int L_B[4], L_X[4], L_Pr[4], L_Po[4];
  for (int l = 0; l < 4; ++l) {
    L_B[l]  = prep(57 + 2 * l, l == 0 ? H_DIM : HIDD, l == 3 ? 1 : HIDD,
                   l == 0 ? 64 : 128, l == 3 ? 16 : 128);
    L_X[l]  = prep(65 + 2 * l, l == 0 ? H_DIM : HIDD, l == 3 ? HO : HIDD,
                   l == 0 ? 64 : 128, l == 3 ? 16 : 128);
    L_Pr[l] = prep(73 + 2 * l, l == 0 ? 2 * H_DIM + (LHC + 1) * E_DIM : HIDD,
                   l == 3 ? 1 : HIDD, l == 0 ? 384 : 128, l == 3 ? 16 : 128);
    L_Po[l] = prep(81 + 2 * l, l == 0 ? H_DIM + 1 : HIDD, l == 3 ? 1 : HIDD,
                   l == 0 ? 96 : 128, l == 3 ? 16 : 128);
  }

  // ---- 1. KNN graph ----
  k_knn<<<dim3(N_PTS / 128), dim3(128), 0, stream>>>(x, srcIdx);

  // ---- 2. node encoder + segment_max + relu ----
  k_build_enc_in<<<dim3(NEDGE * 32 / 256), dim3(256), 0, stream>>>(x, srcIdx, bufIn);
  gemm(L_enc0, bufIn, bufH1, nullptr, NEDGE, 1);
  gemm(L_enc1, bufH1, bufH2, nullptr, NEDGE, 0);                   // msg [E x 64]
  k_segmax_relu<<<dim3(N_PTS * 64 / 256), dim3(256), 0, stream>>>(bufH2, hF, hH);

  // ---- 3. edge encoder (outer relu folded into last layer) ----
  k_build_edge_in<<<dim3(NEDGE * 128 / 256), dim3(256), 0, stream>>>(hH, srcIdx, bufIn);
  gemm(L_ee0, bufIn, bufH1, nullptr, NEDGE, 1);
  gemm(L_ee1, bufH1, eBuf[0], nullptr, NEDGE, 1);

  // ---- 4. LHC interaction layers ----
  for (int li = 0; li < LHC; ++li) {
    k_build_rel_in<<<dim3(NEDGE * 192 / 256), dim3(256), 0, stream>>>(
        hH, eBuf[li], srcIdx, bufIn);
    gemm(L_rel[li][0], bufIn, bufH1, nullptr, NEDGE, 1);
    gemm(L_rel[li][1], bufH1, bufH2, nullptr, NEDGE, 1);
    gemm(L_rel[li][2], bufH2, bufH1, nullptr, NEDGE, 1);
    gemm(L_rel[li][3], bufH1, eBuf[li + 1], nullptr, NEDGE, 0);    // e (no relu)
    k_build_obj_in<<<dim3(N_PTS * 128 / 256), dim3(256), 0, stream>>>(
        hH, eBuf[li + 1], nodeIn);
    gemm(L_obj[li][0], nodeIn, nH1, nullptr, N_PTS, 1);
    gemm(L_obj[li][1], nH1, nH2, nullptr, N_PTS, 1);
    gemm(L_obj[li][2], nH2, nH1, nullptr, N_PTS, 1);
    gemm(L_obj[li][3], nH1, nullptr, deltaF, N_PTS, 0);            // delta_h f32
    k_blend<<<dim3(N_PTS * 64 / 256), dim3(256), 0, stream>>>(hF, deltaF, hH);
  }

  // ---- 5. heads: B (sigmoid), X ----
  gemm(L_B[0], hH, nH1, nullptr, N_PTS, 1);
  gemm(L_B[1], nH1, nH2, nullptr, N_PTS, 1);
  gemm(L_B[2], nH2, nH1, nullptr, N_PTS, 1);
  gemm(L_B[3], nH1, nullptr, bBF, N_PTS, 0);
  k_sigmoid_beta<<<dim3(N_PTS / 256), dim3(256), 0, stream>>>(bBF, out + (size_t)N_PTS * HO);

  gemm(L_X[0], hH, nH1, nullptr, N_PTS, 1);
  gemm(L_X[1], nH1, nH2, nullptr, N_PTS, 1);
  gemm(L_X[2], nH2, nH1, nullptr, N_PTS, 1);
  gemm(L_X[3], nH1, nullptr, xoF, N_PTS, 0);
  k_copy_hout<<<dim3(N_PTS * 8 / 256), dim3(256), 0, stream>>>(xoF, out);

  // ---- 6. P_rel over edges, segment_sum, P_obj ----
  k_build_prel_in<<<dim3(NEDGE * 384 / 256), dim3(256), 0, stream>>>(
      hH, eBuf[0], eBuf[1], eBuf[2], eBuf[3], srcIdx, bufIn);
  gemm(L_Pr[0], bufIn, bufH1, nullptr, NEDGE, 1);
  gemm(L_Pr[1], bufH1, bufH2, nullptr, NEDGE, 1);
  gemm(L_Pr[2], bufH2, bufH1, nullptr, NEDGE, 1);
  gemm(L_Pr[3], bufH1, nullptr, epF, NEDGE, 0);                    // ep [E x 16] f32
  k_build_pobj_in<<<dim3(N_PTS * 96 / 256), dim3(256), 0, stream>>>(hH, epF, nodeIn);
  gemm(L_Po[0], nodeIn, nH1, nullptr, N_PTS, 1);
  gemm(L_Po[1], nH1, nH2, nullptr, N_PTS, 1);
  gemm(L_Po[2], nH2, nH1, nullptr, N_PTS, 1);
  gemm(L_Po[3], nH1, nullptr, trF, N_PTS, 0);
  k_copy_col0<<<dim3(N_PTS / 256), dim3(256), 0, stream>>>(trF, out + (size_t)N_PTS * (HO + 1));
}
